// MinibatchDiscrimination_27058293965069
// MI455X (gfx1250) — matrix operand order repaired and checked by reading, not compiled, not run on hardware
//
#include <hip/hip_runtime.h>

// ---------------------------------------------------------------------------
// MinibatchDiscrimination for MI455X (gfx1250, wave32, WMMA)
//   x: [512, 2048] f32, w: [2048, 1024] f32  ->  out: [512, 32] f32
// Stage 1: h = x@w via v_wmma_f32_16x16x32_bf16, 4 WMMAs per A fragment,
//          B staged in LDS pre-swizzled to fragment order (2x ds_load_b128).
// Stage 2: per output-unit o, all-pairs exp(-L1) reduction, LDS-resident.
// ---------------------------------------------------------------------------

typedef __attribute__((ext_vector_type(16))) __bf16 v16bf;
typedef __attribute__((ext_vector_type(8)))  float  v8f;

#define BATCH 512
#define KDIM  2048
#define NDIM  1024      // UNITS * UNITS_OUT
#define UNITS 32
#define UOUT  32

__device__ __forceinline__ unsigned int pack2_bf16(float lo, float hi) {
    unsigned short a = __builtin_bit_cast(unsigned short, (__bf16)lo);
    unsigned short b = __builtin_bit_cast(unsigned short, (__bf16)hi);
    return ((unsigned int)b << 16) | (unsigned int)a;   // -> v_cvt_pk_bf16_f32
}

// ---------------------------------------------------------------------------
// GEMM: h[BATCH][NDIM] = x[BATCH][KDIM] @ w[KDIM][NDIM]
// Block: 128 threads = 4 waves.  Block tile 64(M) x 64(N), K step 32.
// Wave wv owns rows [tileM+16*wv, +16) and all 64 N columns (4 accumulators),
// so one A fragment feeds 4 WMMAs; the B slab in LDS is shared by all waves.
// Grid: (NDIM/64, BATCH/64) = (16, 8).
// ---------------------------------------------------------------------------
__global__ __launch_bounds__(128)
void mbd_gemm_wmma(const float* __restrict__ x,
                   const float* __restrict__ w,
                   float* __restrict__ h) {
    // B slab 32(K) x 64(N), bf16, pre-swizzled into WMMA B-fragment order:
    // sub-tile nt (16 cols), lane = (((k>>3)&1)<<4)|nn, elem = (k&7)|((k>>4)<<3).
    // Each lane's 16-element fragment is 32 contiguous bytes.
    __shared__ __bf16 shB[4 * 32 * 16];          // 4 KB

    const int tid   = threadIdx.x;
    const int wv    = tid >> 5;        // wave id 0..3
    const int lane  = tid & 31;
    const int hf    = lane >> 4;       // lane half (0/1)
    const int mr    = lane & 15;       // row/col within half
    const int tileN = blockIdx.x * 64;
    const int tileM = blockIdx.y * 64;

    const float* xrow = x + (tileM + wv * 16 + mr) * KDIM;

    v8f acc0 = {}, acc1 = {}, acc2 = {}, acc3 = {};

    for (int kk = 0; kk < KDIM; kk += 32) {
        // ---- cooperative stage of B 32x64 -> bf16, fragment-swizzled -----
        __syncthreads();                        // protect shB from prev iter
        #pragma unroll
        for (int p = 0; p < 8; ++p) {
            const int idx = p * 128 + tid;      // 0..1023 (k-pairs x n)
            const int n   = idx & 63;
            const int k   = (idx >> 6) * 2;     // even k
            const float* src = w + (kk + k) * NDIM + tileN + n;
            const unsigned int pk = pack2_bf16(src[0], src[NDIM]);
            const int nt  = n >> 4;
            const int nn  = n & 15;
            const int off = nt * 512 + (((((k >> 3) & 1) << 4) | nn) << 4) + ((k & 7) | ((k >> 4) << 3));
            *(unsigned int*)&shB[off] = pk;     // dword-aligned (k even)
        }
        __syncthreads();

        // ---- A fragment: ISA 16-bit A 16x32 layout -----------------------
        // lane half 0: K = 0..7 and 16..23 ; half 1: K = 8..15 and 24..31
        float af[16];
        *(float4*)(af + 0)  = *(const float4*)(xrow + kk + hf * 8);
        *(float4*)(af + 4)  = *(const float4*)(xrow + kk + hf * 8 + 4);
        *(float4*)(af + 8)  = *(const float4*)(xrow + kk + 16 + hf * 8);
        *(float4*)(af + 12) = *(const float4*)(xrow + kk + 16 + hf * 8 + 4);
        if (kk + 32 < KDIM)                     // global_prefetch_b8 (CDNA5)
            __builtin_prefetch(xrow + kk + 32 + hf * 8, 0, 1);

        v16bf a;
        #pragma unroll
        for (int e = 0; e < 16; ++e) a[e] = (__bf16)af[e];

        // ---- 4 B fragments: each = 2x ds_load_b128 (contiguous 32B) ------
        const v16bf b0 = *(const v16bf*)&shB[0 * 512 + lane * 16];
        const v16bf b1 = *(const v16bf*)&shB[1 * 512 + lane * 16];
        const v16bf b2 = *(const v16bf*)&shB[2 * 512 + lane * 16];
        const v16bf b3 = *(const v16bf*)&shB[3 * 512 + lane * 16];

        // (neg_a, A, neg_b, B, c_mod, C, reuse_a, reuse_b)
        acc0 = __builtin_amdgcn_wmma_f32_16x16x32_bf16(false, a, false, b0,
                                                       (short)0, acc0, false, false);
        acc1 = __builtin_amdgcn_wmma_f32_16x16x32_bf16(false, a, false, b1,
                                                       (short)0, acc1, false, false);
        acc2 = __builtin_amdgcn_wmma_f32_16x16x32_bf16(false, a, false, b2,
                                                       (short)0, acc2, false, false);
        acc3 = __builtin_amdgcn_wmma_f32_16x16x32_bf16(false, a, false, b3,
                                                       (short)0, acc3, false, false);
    }

    // ---- store C/D: VGPR r -> M = r + 8*hf, N = mr (coalesced 64B runs) --
    const int ibase = tileM + wv * 16 + 8 * hf;
    #pragma unroll
    for (int r = 0; r < 8; ++r) {
        float* dst = h + (ibase + r) * NDIM + tileN + mr;
        dst[0]  = acc0[r];
        dst[16] = acc1[r];
        dst[32] = acc2[r];
        dst[48] = acc3[r];
    }
}

// ---------------------------------------------------------------------------
// Stage 2: one block per output unit o (32 blocks, 512 threads each).
// sh[i*32+u] = h[i, u, o]  (64 KB dynamic LDS).  Thread i keeps its own row
// in registers; inner j-loop does broadcast LDS reads only (conflict-free).
// ---------------------------------------------------------------------------
__global__ __launch_bounds__(512)
void mbd_features(const float* __restrict__ h,
                  float* __restrict__ out) {
    extern __shared__ float sh[];          // BATCH * UNITS floats = 64 KB
    const int o = blockIdx.x;
    const int i = threadIdx.x;

    for (int t = i; t < BATCH * UNITS; t += 512) {
        const int bi = t >> 5;
        const int u  = t & 31;
        sh[t] = h[bi * NDIM + u * UOUT + o];   // L2 lines reused 32x across blocks
    }
    __syncthreads();

    float r[UNITS];
    #pragma unroll
    for (int u = 0; u < UNITS; ++u) r[u] = sh[i * UNITS + u];

    float acc = 0.0f;
    for (int j = 0; j < BATCH; ++j) {
        float d = 0.0f;
        #pragma unroll
        for (int u = 0; u < UNITS; ++u)
            d += __builtin_fabsf(r[u] - sh[j * UNITS + u]);
        acc += __expf(-d);                 // v_exp_f32 (trans, co-executes)
    }

    out[i * UOUT + o] = acc;
}

// ---------------------------------------------------------------------------
extern "C" void kernel_launch(void* const* d_in, const int* in_sizes, int n_in,
                              void* d_out, int out_size, void* d_ws, size_t ws_size,
                              hipStream_t stream) {
    const float* x = (const float*)d_in[0];   // [512, 2048]
    const float* w = (const float*)d_in[1];   // [2048, 1024]
    float* out = (float*)d_out;               // [512, 32]
    float* h   = (float*)d_ws;                // [512, 1024] = 2 MB scratch

    dim3 g1(NDIM / 64, BATCH / 64);           // 16 x 8 blocks, 128 thr each
    mbd_gemm_wmma<<<g1, 128, 0, stream>>>(x, w, h);

    mbd_features<<<32, 512, BATCH * UNITS * sizeof(float), stream>>>(h, out);
}
